// BaseGraphAttNet_79362405696004
// MI455X (gfx1250) — matrix operand order repaired and checked by reading, not compiled, not run on hardware
//
#include <hip/hip_runtime.h>
#include <hip/hip_bf16.h>
#include <stdint.h>

#define BS 8
#define Nn 2048
#define Hh 512
#define NEG_INF (-1.0e9f)
#define SLOPE 0.01f

typedef __attribute__((ext_vector_type(16))) __bf16 v16bf;
typedef __attribute__((ext_vector_type(8)))  float  v8f;
typedef __attribute__((ext_vector_type(4)))  unsigned int u32x4;
typedef __attribute__((ext_vector_type(8)))  int i32x8;
typedef __attribute__((ext_vector_type(4)))  int i32x4;

union Frag16 { v16bf bf; unsigned int u[8]; };

#if __has_builtin(__builtin_amdgcn_tensor_load_to_lds) && __has_builtin(__builtin_amdgcn_s_wait_tensorcnt)
#define HAVE_TDM 1
#else
#define HAVE_TDM 0
#endif

__device__ __forceinline__ unsigned short f2bf(float f) {
    unsigned int u = __float_as_uint(f);
    u += 0x7fffu + ((u >> 16) & 1u);           // round-to-nearest-even
    return (unsigned short)(u >> 16);
}

#if HAVE_TDM
// Build a D# (ISA 8.3/8.4) for a 2-D tile load and issue TENSOR_LOAD_TO_LDS.
// data_size code: 0=1B,1=2B,2=4B,3=8B. Tile rows pack contiguously into LDS.
__device__ __forceinline__ void tdm_load_2d(void* lds, const void* gsrc,
                                            unsigned tile_d0, unsigned tile_d1,
                                            unsigned tensor_d0, unsigned tensor_d1,
                                            unsigned long long stride0,
                                            unsigned ds_code) {
    unsigned long long ga = (unsigned long long)(size_t)gsrc;
    unsigned int la = (unsigned int)(size_t)lds;          // flat LDS addr low 32 = LDS offset
    u32x4 g0;
    g0[0] = 1u;                                           // count=1 (valid), user mode
    g0[1] = la;                                           // lds_addr
    g0[2] = (unsigned)(ga & 0xffffffffu);                 // global_addr[31:0]
    g0[3] = (unsigned)((ga >> 32) & 0x01ffffffu) | (2u << 30);   // addr[56:32] | type=2
    i32x8 g1;
    g1[0] = (int)(ds_code << 16);                         // wg_mask=0 | data_size
    g1[1] = (int)((tensor_d0 & 0xffffu) << 16);           // abar=0 | tdim0[15:0]
    g1[2] = (int)((tensor_d0 >> 16) | ((tensor_d1 & 0xffffu) << 16));
    g1[3] = (int)((tensor_d1 >> 16) | ((tile_d0 & 0xffffu) << 16));
    g1[4] = (int)(tile_d1 & 0xffffu);                     // tile_dim1 | tile_dim2=0
    g1[5] = (int)(stride0 & 0xffffffffu);                 // tensor_dim0_stride[31:0]
    g1[6] = (int)((stride0 >> 32) & 0xffffu);             // stride[47:32] | dim1_stride=0
    g1[7] = 0;
    i32x4 gz = {0, 0, 0, 0};
#if __clang_major__ >= 23
    i32x8 gz8 = {0, 0, 0, 0, 0, 0, 0, 0};
    __builtin_amdgcn_tensor_load_to_lds(g0, g1, gz, gz, gz8, 0);
#else
    __builtin_amdgcn_tensor_load_to_lds(g0, g1, gz, gz, 0);
#endif
}
#endif

// ---------------------------------------------------------------------------
// Kernel 0: fc_w fp32 -> bf16 (512 KB) so GEMM staging is a pure byte move.
// ---------------------------------------------------------------------------
__global__ __launch_bounds__(256)
void wconv_kernel(const float* __restrict__ w, unsigned short* __restrict__ w16) {
    int i = (blockIdx.x * 256 + threadIdx.x) * 4;
    float4 v = *(const float4*)(w + i);
    unsigned int lo = (unsigned)f2bf(v.x) | ((unsigned)f2bf(v.y) << 16);
    unsigned int hi = (unsigned)f2bf(v.z) | ((unsigned)f2bf(v.w) << 16);
    *(uint2*)(w16 + i) = make_uint2(lo, hi);
}

// ---------------------------------------------------------------------------
// Kernel 1: feat_proj = feats @ fc_w^T + fc_b   (bf16 WMMA, fp32 accumulate)
// 256 thr (8 waves): 16 rows x 512 cols. B tile (512x32, stride 512) staged
// by TDM into a DOUBLE buffer: chunk c+1 is in flight while WMMA runs on c.
// ---------------------------------------------------------------------------
__global__ __launch_bounds__(256)
void proj_kernel(const float* __restrict__ feats, const unsigned short* __restrict__ w16,
                 const float* __restrict__ fc_b,
                 float* __restrict__ fp32out, unsigned short* __restrict__ bf16out) {
    __shared__ unsigned short At[16 * 32];         // A tile: [m][k] bf16
    __shared__ unsigned short Bt[2][512 * 32];     // B tile x2: [o][k] bf16

    const int b    = blockIdx.x >> 7;
    const int row0 = (blockIdx.x & 127) << 4;
    const int tid  = threadIdx.x;
    const int lane = tid & 31;
    const int wave = tid >> 5;
    const int kh   = lane >> 4;
    const int nn   = lane & 15;
    const int NC   = Hh / 32;                      // 16 K-chunks

    v8f acc[4] = {};
    const float* fbase = feats + ((size_t)b * Nn + row0) * Hh;

#if HAVE_TDM
    if (tid < 32)                                  // prefetch chunk 0
        tdm_load_2d(Bt[0], w16, /*tile*/32, 512, /*tensor*/Hh, Hh,
                    (unsigned long long)Hh, /*2B*/1u);
#endif

    for (int c = 0; c < NC; ++c) {
        const int kk = c * 32;
        __syncthreads();                           // readers of spare buffer done
        // stage A: 16x32 fp32 -> bf16
        {
            int e = tid * 2;
            int r = e >> 5, cc = e & 31;
            float2 v = *(const float2*)(fbase + (size_t)r * Hh + kk + cc);
            At[r * 32 + cc]     = f2bf(v.x);
            At[r * 32 + cc + 1] = f2bf(v.y);
        }
#if HAVE_TDM
        if (tid < 32) {
            if (c + 1 < NC) {                      // issue next chunk, keep it in flight
                tdm_load_2d(Bt[(c + 1) & 1], w16 + kk + 32, 32, 512, Hh, Hh,
                            (unsigned long long)Hh, 1u);
                __builtin_amdgcn_s_wait_tensorcnt(1);  // oldest (current buf) done
            } else {
                __builtin_amdgcn_s_wait_tensorcnt(0);
            }
        }
#else
        for (int e = tid * 8; e < 512 * 32; e += 256 * 8) {
            int o = e >> 5, cc = e & 31;
            *(u32x4*)&Bt[c & 1][e] = *(const u32x4*)(w16 + (size_t)o * Hh + kk + cc);
        }
#endif
        __syncthreads();

        const unsigned short* Bcur = Bt[c & 1];
        Frag16 A;
        #pragma unroll
        for (int v = 0; v < 8; ++v) {
            int k0 = ((v < 4) ? (2 * v) : (16 + 2 * (v - 4))) + kh * 8;
            A.u[v] = *(const unsigned int*)&At[nn * 32 + k0];
        }
        #pragma unroll
        for (int t = 0; t < 4; ++t) {
            int col = wave * 64 + t * 16 + nn;
            Frag16 Bf;
            #pragma unroll
            for (int v = 0; v < 8; ++v) {
                int k0 = ((v < 4) ? (2 * v) : (16 + 2 * (v - 4))) + kh * 8;
                Bf.u[v] = *(const unsigned int*)&Bcur[col * 32 + k0];
            }
            acc[t] = __builtin_amdgcn_wmma_f32_16x16x32_bf16(
                false, A.bf, false, Bf.bf, (short)0, acc[t], false, false);
        }
    }

    #pragma unroll
    for (int t = 0; t < 4; ++t) {
        int col = wave * 64 + t * 16 + nn;
        float bias = fc_b[col];
        #pragma unroll
        for (int r = 0; r < 8; ++r) {
            int m = r + kh * 8;
            float v = acc[t][r] + bias;
            size_t idx = ((size_t)b * Nn + row0 + m) * Hh + col;
            fp32out[idx] = v;
            bf16out[idx] = f2bf(v);
        }
    }
}

// ---------------------------------------------------------------------------
// Kernel 2: q/k scalar projections; one wave per row, wave32 shuffle reduce
// ---------------------------------------------------------------------------
__global__ __launch_bounds__(256)
void qk_kernel(const float* __restrict__ fp,
               const float* __restrict__ qw, const float* __restrict__ qb,
               const float* __restrict__ kw, const float* __restrict__ kb,
               float* __restrict__ qout, float* __restrict__ kout) {
    const int wave = threadIdx.x >> 5, lane = threadIdx.x & 31;
    const int row  = blockIdx.x * 8 + wave;
    const float* rp = fp + (size_t)row * Hh;
    float sq = 0.f, sk = 0.f;
    for (int h = lane; h < Hh; h += 32) {
        float v = rp[h];
        sq += v * qw[h];
        sk += v * kw[h];
    }
    #pragma unroll
    for (int off = 16; off; off >>= 1) {
        sq += __shfl_xor(sq, off, 32);
        sk += __shfl_xor(sk, off, 32);
    }
    if (lane == 0) { qout[row] = sq + qb[0]; kout[row] = sk + kb[0]; }
}

// ---------------------------------------------------------------------------
// Kernel 3: masked-softmax attention + (atten @ feat_proj) + residual.
// V chunk (32x512 bf16 = contiguous 32 KB) double-buffered via TDM: DMA of
// chunk c+1 overlaps the WMMA accumulation of chunk c.
// ---------------------------------------------------------------------------
__global__ __launch_bounds__(256)
void attn_kernel(const float* __restrict__ feats, const float* __restrict__ adj,
                 const unsigned short* __restrict__ vpro,
                 const float* __restrict__ q, const float* __restrict__ k,
                 float* __restrict__ out) {
    __shared__ float kall[Nn];
    __shared__ float qrow[16];
    __shared__ float rowmax[16];
    __shared__ float rowsum[16];
    __shared__ float redbuf[256];
    __shared__ unsigned short Pt[16 * 32];         // p tile  [m][k]
    __shared__ unsigned short Vt[2][32 * 512];     // V tile x2: [k][col] row-major

    const int b    = blockIdx.x >> 7;
    const int row0 = (blockIdx.x & 127) << 4;
    const int tid  = threadIdx.x;
    const int lane = tid & 31;
    const int wave = tid >> 5;
    const int kh   = lane >> 4;
    const int nn   = lane & 15;
    const int NC   = Nn / 32;                      // 64 K-chunks

    for (int j = tid; j < Nn; j += 256) kall[j] = k[b * Nn + j];
    if (tid < 16) qrow[tid] = q[b * Nn + row0 + tid];
    __syncthreads();

    // ---- pass 1: row maxima of masked+leaky logits ----
    {
        int r = tid >> 4, l = tid & 15;
        const float* arow = adj + ((size_t)b * Nn + row0 + r) * Nn;
        float qv = qrow[r];
        float mx = -3.4e38f;
        for (int j = l; j < Nn; j += 16) {
            __builtin_prefetch(arow + j + 512, 0, 1);     // global_prefetch_b8
            float m = qv + kall[j] + (1.0f - arow[j]) * NEG_INF;
            m = (m >= 0.f) ? m : SLOPE * m;
            mx = fmaxf(mx, m);
        }
        redbuf[tid] = mx;
        __syncthreads();
        if (tid < 16) {
            float m = redbuf[tid * 16];
            #pragma unroll
            for (int l2 = 1; l2 < 16; ++l2) m = fmaxf(m, redbuf[tid * 16 + l2]);
            rowmax[tid] = m;
        }
    }
    __syncthreads();

    v8f acc[4] = {};

    const int rA = tid >> 5, cA = tid & 31;
    float sA = 0.f, sB = 0.f;
    const float* arowA = adj + ((size_t)b * Nn + row0 + rA) * Nn;
    const float* arowB = adj + ((size_t)b * Nn + row0 + rA + 8) * Nn;
    const float qA = qrow[rA],   qB = qrow[rA + 8];
    const float mA = rowmax[rA], mB = rowmax[rA + 8];

    const unsigned short* vbase = vpro + (size_t)b * Nn * Hh;

#if HAVE_TDM
    if (tid < 32)                                  // prefetch chunk 0 (contiguous 32 KB)
        tdm_load_2d(Vt[0], vbase, /*tile*/32 * Hh, 1, /*tensor*/32 * Hh, 1,
                    (unsigned long long)(32 * Hh), /*2B*/1u);
#endif

    for (int c = 0; c < NC; ++c) {
        const int j0 = c * 32;
        __syncthreads();                           // readers of spare buffer done
        // P tile (16x32 unnormalized softmax numerators)
        {
            int j = j0 + cA;
            float kj = kall[j];
            float x = qA + kj + (1.0f - arowA[j]) * NEG_INF;
            x = (x >= 0.f) ? x : SLOPE * x;
            float p = __expf(x - mA);
            sA += p;
            Pt[rA * 32 + cA] = f2bf(p);
            float y = qB + kj + (1.0f - arowB[j]) * NEG_INF;
            y = (y >= 0.f) ? y : SLOPE * y;
            float p2 = __expf(y - mB);
            sB += p2;
            Pt[(rA + 8) * 32 + cA] = f2bf(p2);
        }
#if HAVE_TDM
        if (tid < 32) {
            if (c + 1 < NC) {
                tdm_load_2d(Vt[(c + 1) & 1], vbase + (size_t)(j0 + 32) * Hh,
                            32 * Hh, 1, 32 * Hh, 1,
                            (unsigned long long)(32 * Hh), 1u);
                __builtin_amdgcn_s_wait_tensorcnt(1);  // oldest (current buf) done
            } else {
                __builtin_amdgcn_s_wait_tensorcnt(0);
            }
        }
#else
        {
            const unsigned short* vb = vbase + (size_t)j0 * Hh;
            for (int e = tid * 8; e < 32 * Hh; e += 256 * 8)
                *(u32x4*)&Vt[c & 1][e] = *(const u32x4*)(vb + e);
        }
#endif
        __syncthreads();

        const unsigned short* Vcur = Vt[c & 1];
        Frag16 A;
        #pragma unroll
        for (int v = 0; v < 8; ++v) {
            int k0 = ((v < 4) ? (2 * v) : (16 + 2 * (v - 4))) + kh * 8;
            A.u[v] = *(const unsigned int*)&Pt[nn * 32 + k0];
        }
        #pragma unroll
        for (int t = 0; t < 4; ++t) {
            int col = wave * 64 + t * 16 + nn;
            Frag16 Bf;
            #pragma unroll
            for (int v = 0; v < 8; ++v) {
                int k0 = ((v < 4) ? (2 * v) : (16 + 2 * (v - 4))) + kh * 8;
                unsigned lo = Vcur[k0 * Hh + col];
                unsigned hi = Vcur[(k0 + 1) * Hh + col];
                Bf.u[v] = lo | (hi << 16);
            }
            acc[t] = __builtin_amdgcn_wmma_f32_16x16x32_bf16(
                false, A.bf, false, Bf.bf, (short)0, acc[t], false, false);
        }
    }

    // ---- row sums ----
    redbuf[tid] = sA;
    __syncthreads();
    if (tid < 8) {
        float s = 0.f;
        for (int l2 = 0; l2 < 32; ++l2) s += redbuf[tid * 32 + l2];
        rowsum[tid] = s;
    }
    __syncthreads();
    redbuf[tid] = sB;
    __syncthreads();
    if (tid < 8) {
        float s = 0.f;
        for (int l2 = 0; l2 < 32; ++l2) s += redbuf[tid * 32 + l2];
        rowsum[tid + 8] = s;
    }
    __syncthreads();

    // ---- normalize + residual + store ----
    #pragma unroll
    for (int t = 0; t < 4; ++t) {
        int col = wave * 64 + t * 16 + nn;
        #pragma unroll
        for (int r = 0; r < 8; ++r) {
            int m = r + kh * 8;
            size_t idx = ((size_t)b * Nn + row0 + m) * Hh + col;
            out[idx] = acc[t][r] / rowsum[m] + feats[idx];
        }
    }
}

// ---------------------------------------------------------------------------
extern "C" void kernel_launch(void* const* d_in, const int* in_sizes, int n_in,
                              void* d_out, int out_size, void* d_ws, size_t ws_size,
                              hipStream_t stream) {
    (void)in_sizes; (void)n_in; (void)out_size; (void)ws_size;
    const float* feats = (const float*)d_in[0];
    const float* adj   = (const float*)d_in[1];
    const float* fc_w  = (const float*)d_in[2];
    const float* fc_b  = (const float*)d_in[3];
    const float* q_w   = (const float*)d_in[4];
    const float* q_b   = (const float*)d_in[5];
    const float* k_w   = (const float*)d_in[6];
    const float* k_b   = (const float*)d_in[7];
    float* out = (float*)d_out;

    char* ws = (char*)d_ws;
    const size_t fpBytes = (size_t)BS * Nn * Hh * 4;                     // 32 MB
    float*          fp32 = (float*)ws;
    unsigned short* fp16 = (unsigned short*)(ws + fpBytes);              // 16 MB
    float*          qv   = (float*)(ws + fpBytes + fpBytes / 2);        // 64 KB
    float*          kv   = qv + BS * Nn;                                 // 64 KB
    unsigned short* w16  = (unsigned short*)((char*)(kv + BS * Nn));    // 512 KB

    wconv_kernel<<<(Hh * Hh) / 1024, 256, 0, stream>>>(fc_w, w16);
    proj_kernel <<<BS * (Nn / 16), 256, 0, stream>>>(feats, w16, fc_b, fp32, fp16);
    qk_kernel   <<<(BS * Nn) / 8,  256, 0, stream>>>(fp32, q_w, q_b, k_w, k_b, qv, kv);
    attn_kernel <<<BS * (Nn / 16), 256, 0, stream>>>(feats, adj, fp16, qv, kv, out);
}
